// MultiHeadAttention_69380901699750
// MI455X (gfx1250) — matrix operand order repaired and checked by reading, not compiled, not run on hardware
//
#include <hip/hip_runtime.h>
#include <hip/hip_bf16.h>
#include <stdint.h>

#define DEV __device__ __forceinline__

typedef __attribute__((ext_vector_type(16))) __bf16 bf16x16;
typedef __attribute__((ext_vector_type(8)))  float  f32x8;

union BFrag { bf16x16 v; uint4 q[2]; };

DEV unsigned short f2bf(float f) {
    union { float f; unsigned int u; } x; x.f = f;
    unsigned int r = x.u + 0x7FFFu + ((x.u >> 16) & 1u);   // round-to-nearest-even
    return (unsigned short)(r >> 16);
}
DEV float bf2f(unsigned short u) {
    union { unsigned int u; float f; } x; x.u = ((unsigned int)u) << 16; return x.f;
}

DEV void storeOut(float* p, float v)          { *p = v; }
DEV void storeOut(unsigned short* p, float v) { *p = f2bf(v); }

// Wait until all of this wave's async (global<->LDS) ops have completed.
DEV void waitAsync0() {
#if __has_builtin(__builtin_amdgcn_s_wait_asynccnt)
    __builtin_amdgcn_s_wait_asynccnt(0);
#else
    asm volatile("s_wait_asynccnt 0" ::: "memory");
#endif
}

// One wave-wide async copy: 32 lanes x 32B each (2 x b128 via shared operands,
// INST_OFFSET applies to both LDS and global addresses per CDNA5 ISA §10/15.18).
DEV void asyncCopy32B(unsigned ldsAddr, const void* gAddr) {
    asm volatile(
        "global_load_async_to_lds_b128 %0, %1, off\n\t"
        "global_load_async_to_lds_b128 %0, %1, off offset:16"
        :: "v"(ldsAddr), "v"(gAddr) : "memory");
}

// ---------------------------------------------------------------------------
// fp32 -> bf16 bulk convert (vectorized float4 -> ushort4)
// ---------------------------------------------------------------------------
__global__ __launch_bounds__(256) void cvt_f32_bf16(const float* __restrict__ X,
                                                    unsigned short* __restrict__ Y,
                                                    int n4) {
    int i = blockIdx.x * 256 + threadIdx.x;
    if (i >= n4) return;
    const float4 v = ((const float4*)X)[i];
    ushort4 o;
    o.x = f2bf(v.x); o.y = f2bf(v.y); o.z = f2bf(v.z); o.w = f2bf(v.w);
    ((ushort4*)Y)[i] = o;
}

// ---------------------------------------------------------------------------
// W[n][k] fp32  ->  Wt[k][n] bf16  (tiled LDS transpose, 32x32 tiles)
// ---------------------------------------------------------------------------
__global__ __launch_bounds__(256) void wtrans_bf16(const float* __restrict__ W,
                                                   unsigned short* __restrict__ Wt,
                                                   int N) {
    __shared__ float t[32][33];
    const int tx = threadIdx.x, ty = threadIdx.y;
    const int n0 = blockIdx.y * 32, k0 = blockIdx.x * 32;
#pragma unroll
    for (int i = 0; i < 4; ++i)
        t[ty + 8 * i][tx] = W[(size_t)(n0 + ty + 8 * i) * N + k0 + tx];
    __syncthreads();
#pragma unroll
    for (int i = 0; i < 4; ++i)
        Wt[(size_t)(k0 + ty + 8 * i) * N + n0 + tx] = f2bf(t[tx][ty + 8 * i]);
}

// ---------------------------------------------------------------------------
// Y[M,N] = A[M,K](bf16) @ Bt[K,N](bf16) + bias[N]
// 128x128 block tile, K-step 32, double-buffered LDS filled with
// GLOBAL_LOAD_ASYNC_TO_LDS_B128 (ASYNCcnt), 8 waves (2x4), each wave
// computing 4x2 tiles of v_wmma_f32_16x16x32_bf16.
// ---------------------------------------------------------------------------
template <typename OutT>
__global__ __launch_bounds__(256) void gemm_bf16_wmma(
    const unsigned short* __restrict__ A,
    const unsigned short* __restrict__ Bt,
    const float* __restrict__ bias,
    OutT* __restrict__ Y,
    int M, int N, int K) {
    constexpr int TM = 128, TN = 128, TK = 32;
    constexpr int LDA = TK + 8;   // 40  (pad to kill bank conflicts)
    constexpr int LDB = TN + 8;   // 136
    __shared__ unsigned short As[2][TM * LDA];
    __shared__ unsigned short Bs[2][TK * LDB];

    const int tid  = threadIdx.x;
    const int lane = tid & 31;
    const int wid  = tid >> 5;    // 0..7
    const int wm   = wid >> 2;    // 0..1 : 64-row band
    const int wn   = wid & 3;     // 0..3 : 32-col band
    const int lm   = lane & 15;
    const int lh   = lane >> 4;

    const int m0 = blockIdx.y * TM;
    const int n0 = blockIdx.x * TN;

    // global->LDS staging assignment (32 B per thread per tile)
    const int arow = tid >> 1;          // 0..127
    const int acol = (tid & 1) * 16;    // 0 / 16
    const int brow = tid >> 3;          // 0..31
    const int bcol = (tid & 7) * 16;    // 0..112

    const size_t aBase = (size_t)(m0 + arow) * K + acol;
    const size_t bBase = (size_t)brow * N + n0 + bcol;

    const unsigned ldsA0 = (unsigned)(uintptr_t)&As[0][arow * LDA + acol];
    const unsigned ldsA1 = (unsigned)(uintptr_t)&As[1][arow * LDA + acol];
    const unsigned ldsB0 = (unsigned)(uintptr_t)&Bs[0][brow * LDB + bcol];
    const unsigned ldsB1 = (unsigned)(uintptr_t)&Bs[1][brow * LDB + bcol];

    f32x8 acc[4][2];
#pragma unroll
    for (int i = 0; i < 4; ++i)
#pragma unroll
        for (int j = 0; j < 2; ++j)
#pragma unroll
            for (int r = 0; r < 8; ++r) acc[i][j][r] = 0.0f;

    auto asyncTile = [&](int kt, int buf) {
        asyncCopy32B(buf ? ldsA1 : ldsA0, A + aBase + (size_t)kt * TK);
        asyncCopy32B(buf ? ldsB1 : ldsB0, Bt + bBase + (size_t)kt * TK * N);
    };

    const int nK = K / TK;
    asyncTile(0, 0);

    for (int kt = 0; kt < nK; ++kt) {
        const int cur = kt & 1;
        waitAsync0();          // my async writes into buffer `cur` are done
        __syncthreads();       // everyone's writes are visible
        if (kt + 1 < nK) asyncTile(kt + 1, cur ^ 1);
        if (kt + 2 < nK) {
            __builtin_prefetch(A + aBase + (size_t)(kt + 2) * TK, 0, 1);
            __builtin_prefetch(Bt + bBase + (size_t)(kt + 2) * TK * N, 0, 1);
        }

        // A fragments: lane lm = M row, lh selects K octet (ISA 16-bit A layout)
        BFrag af[4];
#pragma unroll
        for (int i = 0; i < 4; ++i) {
            const unsigned short* s = &As[cur][(wm * 64 + i * 16 + lm) * LDA + 8 * lh];
            af[i].q[0] = *(const uint4*)(s);        // K = 8*lh .. +7
            af[i].q[1] = *(const uint4*)(s + 16);   // K = 16+8*lh .. +7
        }
        // B fragments: lane lm = N column, lh selects K octet (same K order as A)
        BFrag bf_[2];
#pragma unroll
        for (int j = 0; j < 2; ++j) {
            const unsigned short* s = &Bs[cur][wn * 32 + j * 16 + lm];
#pragma unroll
            for (int e = 0; e < 16; ++e)
                bf_[j].v[e] = __builtin_bit_cast(__bf16, s[((e & 7) + 8 * lh + 16 * (e >> 3)) * LDB]);
        }
#pragma unroll
        for (int i = 0; i < 4; ++i)
#pragma unroll
            for (int j = 0; j < 2; ++j)
                acc[i][j] = __builtin_amdgcn_wmma_f32_16x16x32_bf16(
                    false, af[i].v, false, bf_[j].v, (short)0, acc[i][j], false, false);
    }

    // epilogue: D layout -> VGPR r holds M = r + 8*lh, N = lm
#pragma unroll
    for (int j = 0; j < 2; ++j) {
        const int gn = n0 + wn * 32 + j * 16 + lm;
        const float bj = bias[gn];
#pragma unroll
        for (int i = 0; i < 4; ++i) {
            const int gmBase = m0 + wm * 64 + i * 16 + 8 * lh;
#pragma unroll
            for (int r = 0; r < 8; ++r)
                storeOut(&Y[(size_t)(gmBase + r) * N + gn], acc[i][j][r] + bj);
        }
    }
}

// ---------------------------------------------------------------------------
// Attention core: one wave per (b,h).
// scores = q @ (scale*k + rel_h)^T ; softmax ; out = attn @ v  (all WMMA bf16)
// ---------------------------------------------------------------------------
__global__ __launch_bounds__(32) void attn_core(
    const unsigned short* __restrict__ Qb,
    const unsigned short* __restrict__ Kb,
    const unsigned short* __restrict__ Vb,
    const float* __restrict__ rel,          // [63][64]
    unsigned short* __restrict__ Ob) {
    const int bh   = blockIdx.x;
    const int b    = bh >> 5;
    const int h    = bh & 31;
    const int lane = threadIdx.x;
    const int lm   = lane & 15;
    const int lh   = lane >> 4;
    const float scale = 0.125f;             // D^-0.5, D=64

    __shared__ unsigned short k2t[64 * 40]; // [d][j] fused scale*k + rel_h, ld=40
    __shared__ float          sc[32 * 33];  // scores fp32
    __shared__ unsigned short at[32 * 40];  // attn bf16, ld=40

    // build fused K': lane handles column j = lane for all d
    {
        const int j = lane;
        const size_t kbase = ((size_t)(b * 32 + j)) * 2048 + h * 64;
        const int ridx = (h - j + 31) * 64;
        for (int d = 0; d < 64; ++d) {
            float kv = bf2f(Kb[kbase + d]);
            k2t[d * 40 + j] = f2bf(scale * kv + rel[ridx + d]);
        }
    }
    __syncthreads();

    // scores = Q @ K'  (M=32, N=32, K=64 -> 2x2 tiles x 2 k-steps)
    BFrag qa[2][2];
#pragma unroll
    for (int mt = 0; mt < 2; ++mt)
#pragma unroll
        for (int kk = 0; kk < 2; ++kk) {
            const size_t base = ((size_t)(b * 32 + mt * 16 + lm)) * 2048 + h * 64 + kk * 32;
            qa[mt][kk].q[0] = *(const uint4*)&Qb[base + 8 * lh];
            qa[mt][kk].q[1] = *(const uint4*)&Qb[base + 16 + 8 * lh];
        }
    f32x8 s[2][2];
#pragma unroll
    for (int mt = 0; mt < 2; ++mt)
#pragma unroll
        for (int nt = 0; nt < 2; ++nt)
#pragma unroll
            for (int r = 0; r < 8; ++r) s[mt][nt][r] = 0.0f;
#pragma unroll
    for (int nt = 0; nt < 2; ++nt)
#pragma unroll
        for (int kk = 0; kk < 2; ++kk) {
            BFrag kb;
            const unsigned short* p = &k2t[(kk * 32) * 40 + nt * 16 + lm];
#pragma unroll
            for (int e = 0; e < 16; ++e)
                kb.v[e] = __builtin_bit_cast(__bf16, p[((e & 7) + 8 * lh + 16 * (e >> 3)) * 40]);
#pragma unroll
            for (int mt = 0; mt < 2; ++mt)
                s[mt][nt] = __builtin_amdgcn_wmma_f32_16x16x32_bf16(
                    false, qa[mt][kk].v, false, kb.v, (short)0, s[mt][nt], false, false);
        }
#pragma unroll
    for (int mt = 0; mt < 2; ++mt)
#pragma unroll
        for (int nt = 0; nt < 2; ++nt)
#pragma unroll
            for (int r = 0; r < 8; ++r)
                sc[(mt * 16 + r + 8 * lh) * 33 + nt * 16 + lm] = s[mt][nt][r];
    __syncthreads();

    // softmax: lane owns row i = lane
    {
        float* row = &sc[lane * 33];
        float mx = row[0];
        for (int j = 1; j < 32; ++j) mx = fmaxf(mx, row[j]);
        float sum = 0.0f;
        for (int j = 0; j < 32; ++j) { float e = __expf(row[j] - mx); sum += e; row[j] = e; }
        float inv = 1.0f / sum;
        for (int j = 0; j < 32; ++j) at[lane * 40 + j] = f2bf(row[j] * inv);
    }
    __syncthreads();

    // out = attn @ V  (M=32, N=64, K=32)
    BFrag af2[2];
#pragma unroll
    for (int mt = 0; mt < 2; ++mt) {
        const unsigned short* p = &at[(mt * 16 + lm) * 40 + 8 * lh];
        af2[mt].q[0] = *(const uint4*)(p);
        af2[mt].q[1] = *(const uint4*)(p + 16);
    }
#pragma unroll
    for (int nt = 0; nt < 4; ++nt) {
        BFrag vb;
        const size_t vbase = ((size_t)(b * 32)) * 2048 + h * 64 + nt * 16 + lm;
#pragma unroll
        for (int e = 0; e < 16; ++e)
            vb.v[e] = __builtin_bit_cast(__bf16, Vb[vbase + (size_t)((e & 7) + 8 * lh + 16 * (e >> 3)) * 2048]);
#pragma unroll
        for (int mt = 0; mt < 2; ++mt) {
            f32x8 o;
#pragma unroll
            for (int r = 0; r < 8; ++r) o[r] = 0.0f;
            o = __builtin_amdgcn_wmma_f32_16x16x32_bf16(
                false, af2[mt].v, false, vb.v, (short)0, o, false, false);
#pragma unroll
            for (int r = 0; r < 8; ++r) {
                const size_t oi =
                    ((size_t)(b * 32 + mt * 16 + r + 8 * lh)) * 2048 + h * 64 + nt * 16 + lm;
                Ob[oi] = f2bf(o[r]);
            }
        }
    }
}

// ---------------------------------------------------------------------------
extern "C" void kernel_launch(void* const* d_in, const int* in_sizes, int n_in,
                              void* d_out, int out_size, void* d_ws, size_t ws_size,
                              hipStream_t stream) {
    (void)in_sizes; (void)n_in; (void)out_size; (void)ws_size;
    const float* x   = (const float*)d_in[0];
    const float* wq  = (const float*)d_in[1];
    const float* bq  = (const float*)d_in[2];
    const float* wk  = (const float*)d_in[3];
    const float* bk  = (const float*)d_in[4];
    const float* wv  = (const float*)d_in[5];
    const float* bv  = (const float*)d_in[6];
    const float* wo  = (const float*)d_in[7];
    const float* bo  = (const float*)d_in[8];
    const float* rel = (const float*)d_in[9];
    float* out = (float*)d_out;

    constexpr int Bz = 512, Ez = 2048, Hz = 32;
    constexpr int Mz = Bz * 32;                  // 16384 rows
    constexpr size_t XE = (size_t)Mz * Ez;       // 33,554,432 elements
    constexpr size_t WE = (size_t)Ez * Ez;       // 4,194,304 elements

    unsigned char* ws = (unsigned char*)d_ws;
    unsigned short* Xb  = (unsigned short*)ws; ws += XE * 2;
    unsigned short* Wtq = (unsigned short*)ws; ws += WE * 2;
    unsigned short* Wtk = (unsigned short*)ws; ws += WE * 2;
    unsigned short* Wtv = (unsigned short*)ws; ws += WE * 2;
    unsigned short* Wto = (unsigned short*)ws; ws += WE * 2;
    unsigned short* Qb  = (unsigned short*)ws; ws += XE * 2;
    unsigned short* Kb  = (unsigned short*)ws; ws += XE * 2;
    unsigned short* Vb  = (unsigned short*)ws; ws += XE * 2;
    unsigned short* Ob  = Xb;   // reuse X buffer after Q/K/V projections

    cvt_f32_bf16<<<(int)(XE / 4 / 256), 256, 0, stream>>>(x, Xb, (int)(XE / 4));

    dim3 tb(32, 8), tg(Ez / 32, Ez / 32);
    wtrans_bf16<<<tg, tb, 0, stream>>>(wq, Wtq, Ez);
    wtrans_bf16<<<tg, tb, 0, stream>>>(wk, Wtk, Ez);
    wtrans_bf16<<<tg, tb, 0, stream>>>(wv, Wtv, Ez);
    wtrans_bf16<<<tg, tb, 0, stream>>>(wo, Wto, Ez);

    dim3 gg(Ez / 128, Mz / 128);
    gemm_bf16_wmma<unsigned short><<<gg, 256, 0, stream>>>(Xb, Wtq, bq, Qb, Mz, Ez, Ez);
    gemm_bf16_wmma<unsigned short><<<gg, 256, 0, stream>>>(Xb, Wtk, bk, Kb, Mz, Ez, Ez);
    gemm_bf16_wmma<unsigned short><<<gg, 256, 0, stream>>>(Xb, Wtv, bv, Vb, Mz, Ez, Ez);

    attn_core<<<Bz * Hz, 32, 0, stream>>>(Qb, Kb, Vb, rel, Ob);

    gemm_bf16_wmma<float><<<gg, 256, 0, stream>>>(Ob, Wto, bo, out, Mz, Ez, Ez);
}
